// Model_53317724013440
// MI455X (gfx1250) — matrix operand order repaired and checked by reading, not compiled, not run on hardware
//
#include <hip/hip_runtime.h>
#include <hip/hip_bf16.h>

// ---------------------------------------------------------------------------
// MeshGraphNets (EncodeProcessDecode) for MI455X / gfx1250.
// bf16 latents + pre-packed bf16 pair-major weights, v_wmma_f32_16x16x32_bf16
// with f32 accumulate, 64x128 block tile (2x2 16x16 tiles per wave), fused
// bias/ReLU/LayerNorm/residual epilogue, concat inputs gathered on the fly,
// segment_sum via hardware f32 global atomics.
// ---------------------------------------------------------------------------

typedef __attribute__((ext_vector_type(16))) __bf16 v16bf;
typedef __attribute__((ext_vector_type(8)))  float  v8f;

#define TILE_M 64
#define N_NODES 100000
#define N_EDGES 600000
#define LATENT 128
#define NUM_STEPS 15
#define LN_EPS 1e-5f

union AFrag { v16bf v; unsigned u[8]; };
union CFrag { v8f  v; float  e[8];  };

__device__ __forceinline__ unsigned pack_bf16(float lo, float hi)
{
    union { __bf16 b[2]; unsigned u; } t;
    t.b[0] = (__bf16)lo;
    t.b[1] = (__bf16)hi;
    return t.u;
}

__device__ __forceinline__ void unpack_bf16(unsigned u, float& lo, float& hi)
{
    union { unsigned u; __bf16 b[2]; } t;
    t.u = u;
    lo = (float)t.b[0];
    hi = (float)t.b[1];
}

// ---------------------------------------------------------------------------
// Pre-pack f32 weights [Kw,128] into pair-major bf16 dwords:
//   out[kp*128 + c] = pack( W[2kp][c], W[2kp+1][c] ),  rows >= Kw are zero.
// This is exactly the B-fragment layout, so GEMM B staging is a dword copy.
// ---------------------------------------------------------------------------
__global__ void pack_weight_kernel(const float* __restrict__ W,
                                   unsigned* __restrict__ out,
                                   int Kw, int total)
{
    int i = blockIdx.x * 256 + threadIdx.x;
    if (i >= total) return;
    int kp = i >> 7, c = i & 127;
    int kg = 2 * kp;
    float lo = (kg     < Kw) ? W[(long)kg * LATENT + c]       : 0.f;
    float hi = (kg + 1 < Kw) ? W[(long)(kg + 1) * LATENT + c] : 0.f;
    out[i] = pack_bf16(lo, hi);
}

// ---------------------------------------------------------------------------
// Fused MLP layer:  Y = post( A @ W + bias ),  N fixed = 128.
//   MODE 0: A[m][k] = X0[m*KVAL + k]                     (bf16)
//   MODE 1: A = [ Xg[idx0[m]] | Xg[idx1[m]] | X0[m] ]    (edge concat, K=384)
//   MODE 2: A = [ X0[m] | Xf[m] ]                        (node concat, K=256)
// Wp: pre-packed pair-major bf16 [KVAL/2][128] dwords.
// post: RELU, LayerNorm(ln_s,ln_b), +resid — compile-time selected.
// In-place safe for MODE 0 (strictly row-local).
// Block: 256 threads = 8 waves; 64x128 tile; each wave owns 2x2 16x16 tiles.
// ---------------------------------------------------------------------------
template<int KVAL, int MODE, bool RELU, bool LN, bool RESID>
__global__ void __launch_bounds__(256)
mlp_gemm_kernel(const __bf16*   __restrict__ X0,
                const __bf16*   __restrict__ Xg,
                const float*    __restrict__ Xf,
                const int*      __restrict__ idx0,
                const int*      __restrict__ idx1,
                const unsigned* __restrict__ Wp,
                const float*    __restrict__ bias,
                const float*    __restrict__ ln_s,
                const float*    __restrict__ ln_b,
                const __bf16*   __restrict__ resid,
                __bf16*         __restrict__ Y,
                int M)
{
    // LDS tiles hold packed bf16 pairs along K (pair p = elements 2p, 2p+1)
    __shared__ unsigned As[TILE_M][20];     // 64 rows x 16 pairs (pad->80B rows)
    __shared__ unsigned Bs[16][132];        // 16 k-pairs x 128 n (pad: bank split)
    __shared__ float    Cs[TILE_M][LATENT]; // 32 KB
    __shared__ int      rowIdx0[TILE_M], rowIdx1[TILE_M];
    __shared__ float    redS[TILE_M][4], redQ[TILE_M][4];

    const int tid  = threadIdx.x;
    const int lane = tid & 31;
    const int wave = tid >> 5;
    const int m0   = blockIdx.x * TILE_M;

    if (MODE == 1) {
        if (tid < TILE_M) {
            int m = m0 + tid;
            m = (m < M) ? m : (M - 1);
            rowIdx0[tid] = idx0[m];
            rowIdx1[tid] = idx1[m];
        }
    }
    __syncthreads();

    v8f acc00 = {}, acc01 = {}, acc10 = {}, acc11 = {};
    const int mBase = 32 * (wave >> 2);
    const int nBase = 32 * (wave & 3);

    // A loader: each thread stages 8 contiguous bf16 (= 4 pairs = one uint4)
    const int aE = tid * 8;
    const int aR = aE >> 5;                 // tile row 0..63
    const int aC = aE & 31;                 // k offset in chunk {0,8,16,24}
    int aM = m0 + aR;
    aM = (aM < M) ? aM : (M - 1);           // clamp: rows >= M never stored

    // B loader: each thread stages 8 consecutive packed dwords
    const int bKp = (tid * 8) >> 7;         // 0..15
    const int bC  = (tid * 8) & 127;

    #pragma unroll
    for (int kc = 0; kc < KVAL; kc += 32) {
        // ---- stage A tile (64 rows x 32 k -> 16 pairs) ----
        {
            uint4 v;
            if (MODE == 0) {
                v = *(const uint4*)(X0 + (long)aM * KVAL + kc + aC);
            } else if (MODE == 1) {
                if (kc < 128)
                    v = *(const uint4*)(Xg + (long)rowIdx0[aR] * LATENT + kc + aC);
                else if (kc < 256)
                    v = *(const uint4*)(Xg + (long)rowIdx1[aR] * LATENT + (kc - 128) + aC);
                else
                    v = *(const uint4*)(X0 + (long)aM * LATENT + (kc - 256) + aC);
            } else { // MODE 2
                if (kc < 128) {
                    v = *(const uint4*)(X0 + (long)aM * LATENT + kc + aC);
                } else {
                    const float4* fp = (const float4*)(Xf + (long)aM * LATENT + (kc - 128) + aC);
                    float4 f0 = fp[0], f1 = fp[1];
                    v.x = pack_bf16(f0.x, f0.y);
                    v.y = pack_bf16(f0.z, f0.w);
                    v.z = pack_bf16(f1.x, f1.y);
                    v.w = pack_bf16(f1.z, f1.w);
                }
            }
            *(uint4*)&As[aR][aC >> 1] = v;
        }
        // ---- stage B tile: straight dword copy of pre-packed weights ----
        {
            const uint4* wp4 = (const uint4*)(Wp + (long)((kc >> 1) + bKp) * LATENT + bC);
            uint4 w0 = wp4[0], w1 = wp4[1];
            *(uint4*)&Bs[bKp][bC]     = w0;
            *(uint4*)&Bs[bKp][bC + 4] = w1;
        }
        __syncthreads();

        // ---- fragments (ISA 7.12.2 layouts), one dword per VGPR ----
        AFrag a0, a1, b0, b1;
        const int mrow = mBase + (lane & 15);
        const int half = lane >> 4;
        const int n0   = nBase + (lane & 15);
        #pragma unroll
        for (int v = 0; v < 8; ++v) {
            int ai = ((v & 4) ? 8 : 0) + (v & 3) + half * 4;   // A pair index
            a0.u[v] = As[mrow][ai];
            a1.u[v] = As[mrow + 16][ai];
            int kp = ((v & 4) ? 8 : 0) + (v & 3) + half * 4;   // B pair index
            b0.u[v] = Bs[kp][n0];
            b1.u[v] = Bs[kp][n0 + 16];
        }
        acc00 = __builtin_amdgcn_wmma_f32_16x16x32_bf16(false, a0.v, false, b0.v,
                                                        (short)0, acc00, false, false);
        acc01 = __builtin_amdgcn_wmma_f32_16x16x32_bf16(false, a0.v, false, b1.v,
                                                        (short)0, acc01, false, false);
        acc10 = __builtin_amdgcn_wmma_f32_16x16x32_bf16(false, a1.v, false, b0.v,
                                                        (short)0, acc10, false, false);
        acc11 = __builtin_amdgcn_wmma_f32_16x16x32_bf16(false, a1.v, false, b1.v,
                                                        (short)0, acc11, false, false);
        __syncthreads();
    }

    // ---- spill accumulators to LDS (C/D layout: VGPR r -> M = r + 8*half) ----
    {
        CFrag u00, u01, u10, u11;
        u00.v = acc00; u01.v = acc01; u10.v = acc10; u11.v = acc11;
        const int half = lane >> 4;
        const int nn   = nBase + (lane & 15);
        #pragma unroll
        for (int r = 0; r < 8; ++r) {
            int mm = mBase + r + half * 8;
            Cs[mm][nn]           = u00.e[r];
            Cs[mm][nn + 16]      = u01.e[r];
            Cs[mm + 16][nn]      = u10.e[r];
            Cs[mm + 16][nn + 16] = u11.e[r];
        }
    }
    __syncthreads();

    // ---- fused epilogue: 64 rows x 4 threads/row x 32 cols ----
    {
        const int row = tid >> 2;
        const int cb  = (tid & 3) * 32;
        const int m   = m0 + row;
        float vals[32];
        float s = 0.f, q = 0.f;
        #pragma unroll
        for (int j = 0; j < 32; ++j) {
            float y = Cs[row][cb + j] + bias[cb + j];
            if (RELU) y = fmaxf(y, 0.f);
            vals[j] = y; s += y; q += y * y;
        }
        if (LN) {
            redS[row][tid & 3] = s;
            redQ[row][tid & 3] = q;
            __syncthreads();
            float S = 0.f, Q = 0.f;
            #pragma unroll
            for (int j = 0; j < 4; ++j) { S += redS[row][j]; Q += redQ[row][j]; }
            float mu   = S * (1.0f / LATENT);
            float var  = Q * (1.0f / LATENT) - mu * mu;
            float rstd = rsqrtf(var + LN_EPS);
            #pragma unroll
            for (int j = 0; j < 32; ++j)
                vals[j] = (vals[j] - mu) * rstd * ln_s[cb + j] + ln_b[cb + j];
        }
        if (m < M) {
            if (RESID) {
                const uint4* rp = (const uint4*)(resid + (long)m * LATENT + cb);
                #pragma unroll
                for (int g = 0; g < 4; ++g) {
                    uint4 rv = rp[g];
                    unsigned rw[4] = { rv.x, rv.y, rv.z, rv.w };
                    #pragma unroll
                    for (int p = 0; p < 4; ++p) {
                        float lo, hi;
                        unpack_bf16(rw[p], lo, hi);
                        vals[g * 8 + 2 * p]     += lo;
                        vals[g * 8 + 2 * p + 1] += hi;
                    }
                }
            }
            uint4* yp = (uint4*)(Y + (long)m * LATENT + cb);
            #pragma unroll
            for (int g = 0; g < 4; ++g) {
                uint4 o;
                o.x = pack_bf16(vals[g * 8 + 0], vals[g * 8 + 1]);
                o.y = pack_bf16(vals[g * 8 + 2], vals[g * 8 + 3]);
                o.z = pack_bf16(vals[g * 8 + 4], vals[g * 8 + 5]);
                o.w = pack_bf16(vals[g * 8 + 6], vals[g * 8 + 7]);
                yp[g] = o;
            }
        }
    }
}

// ---------------------------------------------------------------------------
// Feature builders (normalized, bf16, K padded to 32 with zeros)
// ---------------------------------------------------------------------------
__global__ void node_feat_kernel(const float* __restrict__ wp,
                                 const float* __restrict__ pwp,
                                 const int*   __restrict__ ntype,
                                 const float* __restrict__ mean,
                                 const float* __restrict__ stdv,
                                 __bf16* __restrict__ out, int N)
{
    int i = blockIdx.x * 256 + threadIdx.x;
    if (i >= N) return;
    float f[12];
    #pragma unroll
    for (int d = 0; d < 3; ++d) f[d] = wp[i * 3 + d] - pwp[i * 3 + d];
    int t = ntype[i];
    #pragma unroll
    for (int d = 0; d < 9; ++d) f[3 + d] = (d == t) ? 1.f : 0.f;
    unsigned pk[16];
    #pragma unroll
    for (int p = 0; p < 6; ++p) {
        float lo = (f[2 * p]     - mean[2 * p])     / stdv[2 * p];
        float hi = (f[2 * p + 1] - mean[2 * p + 1]) / stdv[2 * p + 1];
        pk[p] = pack_bf16(lo, hi);
    }
    #pragma unroll
    for (int p = 6; p < 16; ++p) pk[p] = 0u;
    uint4* op = (uint4*)(out + (long)i * 32);
    op[0] = make_uint4(pk[0],  pk[1],  pk[2],  pk[3]);
    op[1] = make_uint4(pk[4],  pk[5],  pk[6],  pk[7]);
    op[2] = make_uint4(pk[8],  pk[9],  pk[10], pk[11]);
    op[3] = make_uint4(pk[12], pk[13], pk[14], pk[15]);
}

__global__ void edge_feat_kernel(const float* __restrict__ wp,
                                 const float* __restrict__ mp,
                                 const int*   __restrict__ snd,
                                 const int*   __restrict__ rcv,
                                 const float* __restrict__ mean,
                                 const float* __restrict__ stdv,
                                 __bf16* __restrict__ out, int E)
{
    int e = blockIdx.x * 256 + threadIdx.x;
    if (e >= E) return;
    int s = snd[e], r = rcv[e];
    float rw0 = wp[s * 3 + 0] - wp[r * 3 + 0];
    float rw1 = wp[s * 3 + 1] - wp[r * 3 + 1];
    float rw2 = wp[s * 3 + 2] - wp[r * 3 + 2];
    float rm0 = mp[s * 2 + 0] - mp[r * 2 + 0];
    float rm1 = mp[s * 2 + 1] - mp[r * 2 + 1];
    float f[8] = { rw0, rw1, rw2, sqrtf(rw0*rw0 + rw1*rw1 + rw2*rw2),
                   rm0, rm1, sqrtf(rm0*rm0 + rm1*rm1), 0.f };
    unsigned pk[16];
    #pragma unroll
    for (int p = 0; p < 4; ++p) {
        float lo = (p * 2     < 7) ? (f[2 * p]     - mean[2 * p])     / stdv[2 * p]     : 0.f;
        float hi = (p * 2 + 1 < 7) ? (f[2 * p + 1] - mean[2 * p + 1]) / stdv[2 * p + 1] : 0.f;
        pk[p] = pack_bf16(lo, hi);
    }
    #pragma unroll
    for (int p = 4; p < 16; ++p) pk[p] = 0u;
    uint4* op = (uint4*)(out + (long)e * 32);
    op[0] = make_uint4(pk[0],  pk[1],  pk[2],  pk[3]);
    op[1] = make_uint4(pk[4],  pk[5],  pk[6],  pk[7]);
    op[2] = make_uint4(pk[8],  pk[9],  pk[10], pk[11]);
    op[3] = make_uint4(pk[12], pk[13], pk[14], pk[15]);
}

// ---------------------------------------------------------------------------
// segment_sum: one block per edge, 128 lanes = one latent column each.
// ---------------------------------------------------------------------------
__global__ void scatter_add_kernel(const __bf16* __restrict__ edges,
                                   const int* __restrict__ recv,
                                   float* __restrict__ agg, int E)
{
    int e = blockIdx.x;
    if (e >= E) return;
    int c = threadIdx.x;                 // 0..127
    float v = (float)edges[(long)e * LATENT + c];
    atomicAdd(&agg[(long)recv[e] * LATENT + c], v);
}

// ---------------------------------------------------------------------------
// Final decode layer (128 -> 3), un-normalize, integrate positions.
// ---------------------------------------------------------------------------
__global__ void decode_out_kernel(const __bf16* __restrict__ h,
                                  const float* __restrict__ w2,   // [128,3]
                                  const float* __restrict__ b2,   // [3]
                                  const float* __restrict__ omean,
                                  const float* __restrict__ ostd,
                                  const float* __restrict__ wp,
                                  const float* __restrict__ pwp,
                                  float* __restrict__ out, int N)
{
    int i = blockIdx.x * 256 + threadIdx.x;
    if (i >= N) return;
    float a0 = b2[0], a1 = b2[1], a2 = b2[2];
    #pragma unroll 8
    for (int k = 0; k < LATENT; ++k) {
        float x = (float)h[(long)i * LATENT + k];
        a0 += x * w2[k * 3 + 0];
        a1 += x * w2[k * 3 + 1];
        a2 += x * w2[k * 3 + 2];
    }
    float acc[3] = { a0, a1, a2 };
    #pragma unroll
    for (int d = 0; d < 3; ++d) {
        float a = acc[d] * ostd[d] + omean[d];
        out[i * 3 + d] = 2.f * wp[i * 3 + d] + a - pwp[i * 3 + d];
    }
}

// ---------------------------------------------------------------------------
// Host orchestration
// ---------------------------------------------------------------------------
template<int KVAL, int MODE, bool RELU, bool LN, bool RESID>
static inline void run_gemm(hipStream_t s,
                            const __bf16* X0, const __bf16* Xg, const float* Xf,
                            const int* i0, const int* i1,
                            const unsigned* Wp, const float* b,
                            const float* lns, const float* lnb,
                            const __bf16* resid, __bf16* Y, int M)
{
    dim3 grid((M + TILE_M - 1) / TILE_M), block(256);
    mlp_gemm_kernel<KVAL, MODE, RELU, LN, RESID>
        <<<grid, block, 0, s>>>(X0, Xg, Xf, i0, i1, Wp, b, lns, lnb, resid, Y, M);
}

extern "C" void kernel_launch(void* const* d_in, const int* in_sizes, int n_in,
                              void* d_out, int out_size, void* d_ws, size_t ws_size,
                              hipStream_t stream)
{
    const int N = N_NODES, E = N_EDGES;

    const float* world  = (const float*)d_in[0];
    const float* prevw  = (const float*)d_in[1];
    const float* mesh   = (const float*)d_in[2];
    const int*   ntype  = (const int*)  d_in[3];
    const int*   snd    = (const int*)  d_in[4];
    const int*   rcv    = (const int*)  d_in[5];
    const float* nmean  = (const float*)d_in[6];
    const float* nstd   = (const float*)d_in[7];
    const float* emean  = (const float*)d_in[8];
    const float* estd   = (const float*)d_in[9];
    const float* omean  = (const float*)d_in[10];
    const float* ostd   = (const float*)d_in[11];
    auto P = [&](int i) { return (const float*)d_in[i]; };
    // pytree leaf order inside params (dict keys sorted, lists in order):
    //  dec:      bs0=12 bs1=13 bs2=14 ws0=15 ws1=16 ws2=17
    //  enc_edge: bs0=18 bs1=19 bs2=20 ln_b=21 ln_s=22 ws0=23 ws1=24 ws2=25
    //  enc_node: bs0=26 bs1=27 bs2=28 ln_b=29 ln_s=30 ws0=31 ws1=32 ws2=33
    //  proc[s]:  base=34+16*s ; edge {bs0..2, ln_b, ln_s, ws0..2} then node {...}

    // ---- workspace carve-out ----
    char* base = (char*)d_ws;
    size_t off = 0;
    auto alloc = [&](size_t bytes) -> void* {
        void* p = base + off;
        off = (off + bytes + 255) & ~(size_t)255;
        return p;
    };
    __bf16* nodes = (__bf16*)alloc((size_t)N * LATENT * 2);
    __bf16* edges = (__bf16*)alloc((size_t)E * LATENT * 2);
    __bf16* h     = (__bf16*)alloc((size_t)E * LATENT * 2);
    float*  agg   = (float*) alloc((size_t)N * LATENT * 4);
    __bf16* nfeat = (__bf16*)alloc((size_t)N * 32 * 2);
    __bf16* efeat = (__bf16*)alloc((size_t)E * 32 * 2);

    // ---- pre-pack all GEMM weights to pair-major bf16 ----
    auto packW = [&](int pidx, int Kw, int Kpad) -> const unsigned* {
        unsigned* dst = (unsigned*)alloc((size_t)(Kpad / 2) * LATENT * 4);
        int total = (Kpad / 2) * LATENT;
        pack_weight_kernel<<<(total + 255) / 256, 256, 0, stream>>>(P(pidx), dst, Kw, total);
        return dst;
    };
    const unsigned* Wen0 = packW(31, 12, 32);
    const unsigned* Wen1 = packW(32, 128, 128);
    const unsigned* Wen2 = packW(33, 128, 128);
    const unsigned* Wee0 = packW(23, 7, 32);
    const unsigned* Wee1 = packW(24, 128, 128);
    const unsigned* Wee2 = packW(25, 128, 128);
    const unsigned* Wd0  = packW(15, 128, 128);
    const unsigned* Wd1  = packW(16, 128, 128);
    const unsigned* Wproc[NUM_STEPS][6];
    for (int s = 0; s < NUM_STEPS; ++s) {
        int b = 34 + 16 * s;
        Wproc[s][0] = packW(b + 5,  384, 384);
        Wproc[s][1] = packW(b + 6,  128, 128);
        Wproc[s][2] = packW(b + 7,  128, 128);
        Wproc[s][3] = packW(b + 13, 256, 256);
        Wproc[s][4] = packW(b + 14, 128, 128);
        Wproc[s][5] = packW(b + 15, 128, 128);
    }

    // ---- build features ----
    node_feat_kernel<<<(N + 255) / 256, 256, 0, stream>>>(world, prevw, ntype,
                                                          nmean, nstd, nfeat, N);
    edge_feat_kernel<<<(E + 255) / 256, 256, 0, stream>>>(world, mesh, snd, rcv,
                                                          emean, estd, efeat, E);

    // ---- encode nodes: [12->128]+relu, [128->128]+relu, [128->128]+LN ----
    run_gemm<32, 0, true, false, false>(stream, nfeat, 0, 0, 0, 0, Wen0, P(26), 0, 0, 0, h, N);
    run_gemm<128, 0, true, false, false>(stream, h,    0, 0, 0, 0, Wen1, P(27), 0, 0, 0, h, N);
    run_gemm<128, 0, false, true, false>(stream, h,    0, 0, 0, 0, Wen2, P(28), P(30), P(29), 0, nodes, N);

    // ---- encode edges ----
    run_gemm<32, 0, true, false, false>(stream, efeat, 0, 0, 0, 0, Wee0, P(18), 0, 0, 0, h, E);
    run_gemm<128, 0, true, false, false>(stream, h,    0, 0, 0, 0, Wee1, P(19), 0, 0, 0, h, E);
    run_gemm<128, 0, false, true, false>(stream, h,    0, 0, 0, 0, Wee2, P(20), P(22), P(21), 0, edges, E);

    // ---- 15 processor steps ----
    for (int s = 0; s < NUM_STEPS; ++s) {
        int b = 34 + 16 * s;
        // edge MLP on [nodes[snd] | nodes[rcv] | edges]  (K=384), residual
        run_gemm<384, 1, true, false, false>(stream, edges, nodes, 0, snd, rcv,
                                             Wproc[s][0], P(b + 0), 0, 0, 0, h, E);
        run_gemm<128, 0, true, false, false>(stream, h, 0, 0, 0, 0,
                                             Wproc[s][1], P(b + 1), 0, 0, 0, h, E);
        run_gemm<128, 0, false, true, true>(stream, h, 0, 0, 0, 0,
                                            Wproc[s][2], P(b + 2), P(b + 4), P(b + 3),
                                            edges, edges, E);
        // aggregate updated edges into nodes
        hipMemsetAsync(agg, 0, (size_t)N * LATENT * 4, stream);
        scatter_add_kernel<<<E, 128, 0, stream>>>(edges, rcv, agg, E);
        // node MLP on [nodes | agg]  (K=256), residual
        run_gemm<256, 2, true, false, false>(stream, nodes, 0, agg, 0, 0,
                                             Wproc[s][3], P(b + 8), 0, 0, 0, h, N);
        run_gemm<128, 0, true, false, false>(stream, h, 0, 0, 0, 0,
                                             Wproc[s][4], P(b + 9), 0, 0, 0, h, N);
        run_gemm<128, 0, false, true, true>(stream, h, 0, 0, 0, 0,
                                            Wproc[s][5], P(b + 10), P(b + 12), P(b + 11),
                                            nodes, nodes, N);
    }

    // ---- decode: [128->128]+relu, [128->128]+relu, [128->3] + integrate ----
    run_gemm<128, 0, true, false, false>(stream, nodes, 0, 0, 0, 0, Wd0, P(12), 0, 0, 0, h, N);
    run_gemm<128, 0, true, false, false>(stream, h,     0, 0, 0, 0, Wd1, P(13), 0, 0, 0, h, N);
    decode_out_kernel<<<(N + 255) / 256, 256, 0, stream>>>(h, P(17), P(14), omean, ostd,
                                                           world, prevw, (float*)d_out, N);
}
